// VecAwCell_26173530702403
// MI455X (gfx1250) — matrix operand order repaired and checked by reading, not compiled, not run on hardware
//
#include <hip/hip_runtime.h>
#include <math.h>

// ---------------------------------------------------------------------------
// VecAwCell for MI455X (gfx1250): bf16 WMMA GEMMs, weights pre-swizzled into
// B-fragment order and kept L2-resident; 256 sequential steps as graph nodes.
// ---------------------------------------------------------------------------

typedef __attribute__((ext_vector_type(16))) __bf16 v16bf;
typedef __attribute__((ext_vector_type(8)))  __bf16 v8bf;
typedef __attribute__((ext_vector_type(8)))  float  v8f;

#define B_DIM 64
#define U_DIM 1024
#define F_DIM 512
#define T_DIM 256
#define NT_COUNT 64              // U_DIM / 16 n-tiles (all weights have N=1024)

// ---------------------------- WMMA GEMM core -------------------------------

__device__ __forceinline__ v16bf combine8(v8bf lo, v8bf hi) {
  v16bf a;
#pragma unroll
  for (int i = 0; i < 8; ++i) { a[i] = lo[i]; a[i + 8] = hi[i]; }
  return a;
}

// Accumulate one 16x16 tile: C(mt,nt) += A[64 x 32*kChunks] @ B, where B is
// pre-swizzled: tile (kc,nt) stored as 32 lanes x 16 bf16 contiguous.
__device__ __forceinline__ v8f tile_accum(v8f acc,
                                          const __bf16* __restrict__ A, int lda,
                                          int kChunks,
                                          const __bf16* __restrict__ Bsw,
                                          int nt, int mt) {
  const int lane = threadIdx.x & 31;
  const int row  = mt * 16 + (lane & 15);
  const int hi8  = (lane & 16) ? 8 : 0;   // A lanes 16-31 take K offset +8
  for (int kc = 0; kc < kChunks; ++kc) {
    // B fragment: lane-contiguous 32B (two global_load_b128)
    v16bf bfrag = *(const v16bf*)(Bsw + (((size_t)kc * NT_COUNT + nt) * 32 + lane) * 16);
    // A fragment: row-major, elements e0..7 at K=base, e8..15 at K=base+16
    const __bf16* ap = A + (size_t)row * lda + kc * 32 + hi8;
    v16bf afrag = combine8(*(const v8bf*)ap, *(const v8bf*)(ap + 16));
    acc = __builtin_amdgcn_wmma_f32_16x16x32_bf16(false, afrag, false, bfrag,
                                                  (short)0, acc, false, false);
  }
  return acc;
}

__device__ __forceinline__ void store_tile(v8f acc, float* __restrict__ C,
                                           __bf16* __restrict__ Cb,
                                           int nt, int mt) {
  const int lane = threadIdx.x & 31;
  const int col  = nt * 16 + (lane & 15);
  const int rhi  = (lane & 16) ? 8 : 0;   // C/D layout: lanes 16-31 -> M+8
#pragma unroll
  for (int r = 0; r < 8; ++r) {
    const int row = mt * 16 + r + rhi;
    const float v = acc[r];
    C[row * U_DIM + col] = v;
    if (Cb) Cb[row * U_DIM + col] = (__bf16)v;
  }
}

__device__ __forceinline__ v8f zero_acc() {
  v8f z;
#pragma unroll
  for (int i = 0; i < 8; ++i) z[i] = 0.0f;
  return z;
}

// Generic C[sel] = A @ W[sel]; grid = (32, nmat), block = 256 (8 waves).
__global__ void gemm64_kernel(const __bf16* __restrict__ A, int lda, int kChunks,
                              const __bf16* B0, const __bf16* B1, const __bf16* B2,
                              float* C0, float* C1, float* C2,
                              __bf16* Cb0, __bf16* Cb1, __bf16* Cb2) {
  const int wave = threadIdx.x >> 5;
  const int tile = blockIdx.x * 8 + wave;   // 0..255
  const int mt = tile & 3;
  const int nt = tile >> 2;
  const __bf16* B = (blockIdx.y == 0) ? B0 : (blockIdx.y == 1) ? B1 : B2;
  float*        C = (blockIdx.y == 0) ? C0 : (blockIdx.y == 1) ? C1 : C2;
  __bf16*      Cb = (blockIdx.y == 0) ? Cb0 : (blockIdx.y == 1) ? Cb1 : Cb2;
  v8f acc = zero_acc();
  acc = tile_accum(acc, A, lda, kChunks, B, nt, mt);
  store_tile(acc, C, Cb, nt, mt);
}

// xx = x_t @ Kx + h_prev @ Kh ; grid = (32,1)
__global__ void gemm_xx_kernel(const __bf16* __restrict__ xb_t,   // lda = T*F
                               const __bf16* __restrict__ Kxs,
                               const __bf16* __restrict__ hb,     // lda = U
                               const __bf16* __restrict__ Khs,
                               float* __restrict__ xx, __bf16* __restrict__ xxb) {
  const int wave = threadIdx.x >> 5;
  const int tile = blockIdx.x * 8 + wave;
  const int mt = tile & 3;
  const int nt = tile >> 2;
  v8f acc = zero_acc();
  acc = tile_accum(acc, xb_t, T_DIM * F_DIM, F_DIM / 32, Kxs, nt, mt);
  acc = tile_accum(acc, hb,   U_DIM,         U_DIM / 32, Khs, nt, mt);
  store_tile(acc, xx, xxb, nt, mt);
}

// ------------------------- elementwise / LN / attn -------------------------

__device__ __forceinline__ float sigmoidf_(float z) { return 1.0f / (1.0f + expf(-z)); }

__device__ float block_reduce(float v, float* red) {
  const int tid = threadIdx.x;
  red[tid] = v;
  __syncthreads();
  for (int s = 128; s > 0; s >>= 1) {
    if (tid < s) red[tid] += red[tid + s];
    __syncthreads();
  }
  float r = red[0];
  __syncthreads();
  return r;
}

// One block per batch row: gates, cell update, swish chain, two LayerNorms.
__global__ void cell_kernel(const float* __restrict__ xx,
                            const float* __restrict__ gfr,
                            const float* __restrict__ gir,
                            float* __restrict__ c,            // in: c_prev, out: LN(longterm)
                            const float* __restrict__ g1, const float* __restrict__ b1,
                            const float* __restrict__ g2, const float* __restrict__ b2,
                            float* __restrict__ sln, __bf16* __restrict__ slnb) {
  __shared__ float red[256];
  const int b = blockIdx.x, tid = threadIdx.x;
  const int base = b * U_DIM;
  float st[4], lt[4];
#pragma unroll
  for (int j = 0; j < 4; ++j) {
    const int u = j * 256 + tid;
    const float z = xx[base + u];
    const float f = sigmoidf_(gfr[base + u]);
    const float i = sigmoidf_(gir[base + u]);
    const float cand = tanhf(z);
    const float lterm = f * c[base + u] + i * cand;
    const float s = lterm + z * sigmoidf_(z);
    st[j] = s * sigmoidf_(s);
    lt[j] = lterm;
  }
  // LN(shortterm) -> sln (+bf16 copy for the q/k/v GEMM)
  {
    float s1 = st[0] + st[1] + st[2] + st[3];
    const float mean = block_reduce(s1, red) * (1.0f / U_DIM);
    float s2 = 0.f;
#pragma unroll
    for (int j = 0; j < 4; ++j) { const float d = st[j] - mean; s2 += d * d; }
    const float var = block_reduce(s2, red) * (1.0f / U_DIM);
    const float r = rsqrtf(var + 1e-3f);
#pragma unroll
    for (int j = 0; j < 4; ++j) {
      const int u = j * 256 + tid;
      const float y = (st[j] - mean) * r * g1[u] + b1[u];
      sln[base + u] = y;
      slnb[base + u] = (__bf16)y;
    }
  }
  // LN(longterm) -> c (carried to next step)
  {
    float s1 = lt[0] + lt[1] + lt[2] + lt[3];
    const float mean = block_reduce(s1, red) * (1.0f / U_DIM);
    float s2 = 0.f;
#pragma unroll
    for (int j = 0; j < 4; ++j) { const float d = lt[j] - mean; s2 += d * d; }
    const float var = block_reduce(s2, red) * (1.0f / U_DIM);
    const float r = rsqrtf(var + 1e-3f);
#pragma unroll
    for (int j = 0; j < 4; ++j) {
      const int u = j * 256 + tid;
      c[base + u] = (lt[j] - mean) * r * g2[u] + b2[u];
    }
  }
}

// 8-head, 8x8-score attention per batch element. One block per b, wave per head.
__global__ void attn_kernel(const float* __restrict__ q, const float* __restrict__ k,
                            const float* __restrict__ v, __bf16* __restrict__ att) {
  const int b = blockIdx.x;
  const int wave = threadIdx.x >> 5, lane = threadIdx.x & 31;
  const int h = wave;
  const float scale = 0.08838834764831845f;  // 1/sqrt(128)
  const int base = b * U_DIM + h * 128;
  float ql[4];
#pragma unroll
  for (int j = 0; j < 4; ++j) ql[j] = q[base + j * 32 + lane];
  float s[8];
#pragma unroll
  for (int g = 0; g < 8; ++g) {
    const int kb = b * U_DIM + g * 128;
    float p = 0.f;
#pragma unroll
    for (int j = 0; j < 4; ++j) p += ql[j] * k[kb + j * 32 + lane];
#pragma unroll
    for (int off = 16; off > 0; off >>= 1) p += __shfl_xor(p, off, 32);
    s[g] = p * scale;
  }
  float mx = s[0];
#pragma unroll
  for (int g = 1; g < 8; ++g) mx = fmaxf(mx, s[g]);
  float sum = 0.f;
#pragma unroll
  for (int g = 0; g < 8; ++g) { s[g] = expf(s[g] - mx); sum += s[g]; }
  const float inv = 1.0f / sum;
#pragma unroll
  for (int j = 0; j < 4; ++j) {
    float a = 0.f;
#pragma unroll
    for (int g = 0; g < 8; ++g) a += s[g] * inv * v[b * U_DIM + g * 128 + j * 32 + lane];
    att[base + j * 32 + lane] = (__bf16)a;
  }
}

// out[b,t,:] = sln + attended@Wo ; also carries h_next (bf16) for next step.
__global__ void out_kernel(const float* __restrict__ sln, const float* __restrict__ ao,
                           float* __restrict__ out, __bf16* __restrict__ hb, int t) {
  const int i = blockIdx.x * 256 + threadIdx.x;  // 0..65535
  const int b = i >> 10, u = i & 1023;
  const float val = sln[i] + ao[i];
  out[((size_t)b * T_DIM + t) * U_DIM + u] = val;
  hb[i] = (__bf16)val;
}

// ------------------------------- converts ----------------------------------

__global__ void cvt_bf16_kernel(const float* __restrict__ s, __bf16* __restrict__ d, int n) {
  const int i = blockIdx.x * 256 + threadIdx.x;
  if (i < n) d[i] = (__bf16)s[i];
}

// fp32 weight [K x 1024] -> bf16 in WMMA B-fragment order:
// tile (kc,nt): 32 lanes x 16 elems contiguous; lane L covers N=nt*16+(L&15),
// K = kc*32 + (e&7) + (L>=16 ? 8 : 0) + (e>=8 ? 16 : 0).
__global__ void cvt_swz_kernel(const float* __restrict__ W, __bf16* __restrict__ D) {
  const int i = blockIdx.x * 256 + threadIdx.x;
  const int e    = i & 15;
  const int lane = (i >> 4) & 31;
  const int nt   = (i >> 9) & 63;
  const int kc   = i >> 15;
  const int kk = kc * 32 + (e & 7) + ((lane & 16) >> 1) + ((e & 8) << 1);
  const int nn = nt * 16 + (lane & 15);
  D[i] = (__bf16)W[(size_t)kk * U_DIM + nn];
}

__global__ void init_state_kernel(__bf16* __restrict__ hb, float* __restrict__ c) {
  const int i = blockIdx.x * 256 + threadIdx.x;  // 65536
  hb[i] = (__bf16)0.0f;
  c[i] = 0.0f;
}

// ------------------------------- launcher ----------------------------------

extern "C" void kernel_launch(void* const* d_in, const int* in_sizes, int n_in,
                              void* d_out, int out_size, void* d_ws, size_t ws_size,
                              hipStream_t stream) {
  const float* x  = (const float*)d_in[0];
  const float* Kx = (const float*)d_in[1];
  const float* Kh = (const float*)d_in[2];
  const float* Wq = (const float*)d_in[3];
  const float* Wk = (const float*)d_in[4];
  const float* Wv = (const float*)d_in[5];
  const float* Wo = (const float*)d_in[6];
  const float* Wf = (const float*)d_in[7];
  const float* Wi = (const float*)d_in[8];
  const float* g1 = (const float*)d_in[9];
  const float* b1 = (const float*)d_in[10];
  const float* g2 = (const float*)d_in[11];
  const float* b2 = (const float*)d_in[12];
  float* out = (float*)d_out;

  char* ws = (char*)d_ws;
  size_t off = 0;
  auto alloc = [&](size_t bytes) -> char* {
    char* p = ws + off;
    off = (off + bytes + 255) & ~(size_t)255;
    return p;
  };

  __bf16* xb   = (__bf16*)alloc((size_t)B_DIM * T_DIM * F_DIM * 2);  // 16 MB
  __bf16* Kxs  = (__bf16*)alloc((size_t)F_DIM * U_DIM * 2);
  __bf16* Khs  = (__bf16*)alloc((size_t)U_DIM * U_DIM * 2);
  __bf16* Wqs  = (__bf16*)alloc((size_t)U_DIM * U_DIM * 2);
  __bf16* Wks  = (__bf16*)alloc((size_t)U_DIM * U_DIM * 2);
  __bf16* Wvs  = (__bf16*)alloc((size_t)U_DIM * U_DIM * 2);
  __bf16* Wos  = (__bf16*)alloc((size_t)U_DIM * U_DIM * 2);
  __bf16* Wfs  = (__bf16*)alloc((size_t)U_DIM * U_DIM * 2);
  __bf16* Wis  = (__bf16*)alloc((size_t)U_DIM * U_DIM * 2);
  __bf16* hb   = (__bf16*)alloc((size_t)B_DIM * U_DIM * 2);
  float*  cst  = (float*)alloc((size_t)B_DIM * U_DIM * 4);
  float*  xx   = (float*)alloc((size_t)B_DIM * U_DIM * 4);
  __bf16* xxb  = (__bf16*)alloc((size_t)B_DIM * U_DIM * 2);
  float*  gf   = (float*)alloc((size_t)B_DIM * U_DIM * 4);
  float*  gi   = (float*)alloc((size_t)B_DIM * U_DIM * 4);
  float*  sln  = (float*)alloc((size_t)B_DIM * U_DIM * 4);
  __bf16* slnb = (__bf16*)alloc((size_t)B_DIM * U_DIM * 2);
  float*  qq   = (float*)alloc((size_t)B_DIM * U_DIM * 4);
  float*  kk   = (float*)alloc((size_t)B_DIM * U_DIM * 4);
  float*  vv   = (float*)alloc((size_t)B_DIM * U_DIM * 4);
  __bf16* attb = (__bf16*)alloc((size_t)B_DIM * U_DIM * 2);
  float*  ao   = (float*)alloc((size_t)B_DIM * U_DIM * 4);

  // One-time converts: x -> bf16; weights -> bf16 B-fragment order (L2-resident).
  cvt_bf16_kernel<<<(B_DIM * T_DIM * F_DIM) / 256, 256, 0, stream>>>(
      x, xb, B_DIM * T_DIM * F_DIM);
  cvt_swz_kernel<<<(F_DIM * U_DIM) / 256, 256, 0, stream>>>(Kx, Kxs);
  cvt_swz_kernel<<<(U_DIM * U_DIM) / 256, 256, 0, stream>>>(Kh, Khs);
  cvt_swz_kernel<<<(U_DIM * U_DIM) / 256, 256, 0, stream>>>(Wq, Wqs);
  cvt_swz_kernel<<<(U_DIM * U_DIM) / 256, 256, 0, stream>>>(Wk, Wks);
  cvt_swz_kernel<<<(U_DIM * U_DIM) / 256, 256, 0, stream>>>(Wv, Wvs);
  cvt_swz_kernel<<<(U_DIM * U_DIM) / 256, 256, 0, stream>>>(Wo, Wos);
  cvt_swz_kernel<<<(U_DIM * U_DIM) / 256, 256, 0, stream>>>(Wf, Wfs);
  cvt_swz_kernel<<<(U_DIM * U_DIM) / 256, 256, 0, stream>>>(Wi, Wis);
  init_state_kernel<<<(B_DIM * U_DIM) / 256, 256, 0, stream>>>(hb, cst);

  for (int t = 0; t < T_DIM; ++t) {
    // xx = x_t @ Kx + h_prev @ Kh
    gemm_xx_kernel<<<dim3(32, 1), 256, 0, stream>>>(
        xb + (size_t)t * F_DIM, Kxs, hb, Khs, xx, xxb);
    // f/i gate pre-activations (fused pair)
    gemm64_kernel<<<dim3(32, 2), 256, 0, stream>>>(
        xxb, U_DIM, U_DIM / 32, Wfs, Wis, nullptr,
        gf, gi, nullptr, nullptr, nullptr, nullptr);
    // cell update + swish chain + two LayerNorms
    cell_kernel<<<B_DIM, 256, 0, stream>>>(xx, gf, gi, cst, g1, b1, g2, b2, sln, slnb);
    // q,k,v (fused triple)
    gemm64_kernel<<<dim3(32, 3), 256, 0, stream>>>(
        slnb, U_DIM, U_DIM / 32, Wqs, Wks, Wvs,
        qq, kk, vv, nullptr, nullptr, nullptr);
    // per-batch 8-head attention over heads axis
    attn_kernel<<<B_DIM, 256, 0, stream>>>(qq, kk, vv, attb);
    // attended @ W_o
    gemm64_kernel<<<dim3(32, 1), 256, 0, stream>>>(
        attb, U_DIM, U_DIM / 32, Wos, nullptr, nullptr,
        ao, nullptr, nullptr, nullptr, nullptr, nullptr);
    // residual + write ys[:,t,:] + carry h
    out_kernel<<<(B_DIM * U_DIM) / 256, 256, 0, stream>>>(sln, ao, out, hb, t);
  }
}
